// CachedEHREmbeddings_74947179315384
// MI455X (gfx1250) — matrix operand order repaired and checked by reading, not compiled, not run on hardware
//
#include <hip/hip_runtime.h>

typedef __attribute__((ext_vector_type(16))) _Float16 v16h;
typedef __attribute__((ext_vector_type(8)))  _Float16 v8h;
typedef __attribute__((ext_vector_type(4)))  _Float16 v4h;
typedef __attribute__((ext_vector_type(8)))  float    v8f;

namespace {
constexpr int Ss = 2048;              // sequence length
constexpr int Hh = 768;               // hidden
constexpr int Tt = 32;                // sinusoid dims
constexpr int KD = Hh + 2 * Tt;       // 832 = GEMM K, 26 tiles of 32
constexpr int KT = KD / 32;           // 26
constexpr int NT = Hh / 16;           // 48 N-tiles; 6 per wave (8 waves)
constexpr int APAD = KD + 8;          // 840 f16 (1680 B row: 16B-aligned, bank-skewed)
constexpr int OPAD = Hh + 8;          // 776 f16
}

// Pre-transpose lin_W [K=832][N=768] f32 -> wT [N][K] f16 (coalesced writes).
__global__ __launch_bounds__(256)
void ehr_prep_wT(const float* __restrict__ lin_W, _Float16* __restrict__ wT) {
  int idx = blockIdx.x * 256 + threadIdx.x;
  if (idx >= Hh * KD) return;
  int n = idx / KD;
  int k = idx - n * KD;
  wT[idx] = (_Float16)lin_W[(size_t)k * Hh + n];
}

__global__ __launch_bounds__(256)
void ehr_fused(const int* __restrict__ input_ids, const int* __restrict__ type_ids,
               const float* __restrict__ time_stamps, const float* __restrict__ ages,
               const int* __restrict__ visit_orders, const int* __restrict__ visit_segments,
               const float* __restrict__ W_word, const float* __restrict__ W_type,
               const float* __restrict__ W_order, const float* __restrict__ W_seg,
               const float* __restrict__ time_w, const float* __restrict__ time_phi,
               const float* __restrict__ age_w, const float* __restrict__ age_phi,
               const _Float16* __restrict__ wT, const float* __restrict__ lin_b,
               const float* __restrict__ ln_g, const float* __restrict__ ln_beta,
               float* __restrict__ out)
{
  __shared__ _Float16 Atile[16 * APAD];   // fused input tile, f16
  __shared__ _Float16 Otile[16 * OPAD];   // tanh(GEMM) tile, f16
  __shared__ float rsum[256];
  __shared__ float rsq[256];

  const int t = threadIdx.x;
  const int token0 = blockIdx.x * 16;     // 16 consecutive tokens per block

  // ---- Phase 1a: word-embedding gather (f32 -> f16 into LDS), float4 coalesced ----
  #pragma unroll
  for (int it = 0; it < 12; ++it) {
    int f   = t + it * 256;               // 0..3071 over 16 rows x 192 float4
    int row = f / 192;
    int c4  = f - row * 192;
    int id  = input_ids[token0 + row];
    float4 w = ((const float4*)(W_word + (size_t)id * Hh))[c4];
    v4h d;
    d[0] = (_Float16)w.x; d[1] = (_Float16)w.y;
    d[2] = (_Float16)w.z; d[3] = (_Float16)w.w;
    *(v4h*)(Atile + row * APAD + c4 * 4) = d;
  }
  // ---- Phase 1b: sinusoidal time-delta / age features (cols 768..831) ----
  {
    int row = t >> 4;
    int j0  = (t & 15) * 4;
    int tok = token0 + row;
    float dt  = ((tok & (Ss - 1)) == 0) ? 0.f
              : (time_stamps[tok] - time_stamps[tok - 1]);
    float age = ages[tok];
    #pragma unroll
    for (int u = 0; u < 4; ++u) {
      int j = j0 + u;
      float val = (j < Tt) ? sinf(dt  * time_w[j]       + time_phi[j])
                           : sinf(age * age_w[j - Tt]   + age_phi[j - Tt]);
      Atile[row * APAD + Hh + j] = (_Float16)val;
    }
  }
  __syncthreads();

  // ---- Phase 2: GEMM via v_wmma_f32_16x16x32_f16 ----
  const int wid  = t >> 5;
  const int lane = t & 31;
  const int nl   = lane & 15;             // N within tile / M row for A
  const int hi   = lane >> 4;             // lane-half
  const _Float16* arow = Atile + nl * APAD + hi * 8;
  for (int nt = 0; nt < NT / 8; ++nt) {   // 6 N-tiles per wave
    const int n0 = (wid * (NT / 8) + nt) * 16;
    const _Float16* bcol = wT + (size_t)(n0 + nl) * KD + hi * 8;
    v8f acc = {};
    for (int kt = 0; kt < KT; ++kt) {
      const int k0 = kt * 32;
      // A frag: lanes hold M=lane&15; K runs [k0+hi*8, +8) and [k0+16+hi*8, +8)
      v8h alo = *(const v8h*)(arow + k0);
      v8h ahi = *(const v8h*)(arow + k0 + 16);
      v16h a;
      #pragma unroll
      for (int i = 0; i < 8; ++i) { a[i] = alo[i]; a[i + 8] = ahi[i]; }
      // B frag: lane holds N=lane&15; K runs [k0+hi*8, +8) and [k0+16+hi*8, +8)
      v8h blo = *(const v8h*)(bcol + k0);
      v8h bhi = *(const v8h*)(bcol + k0 + 16);
      v16h b;
      #pragma unroll
      for (int i = 0; i < 8; ++i) { b[i] = blo[i]; b[i + 8] = bhi[i]; }
      acc = __builtin_amdgcn_wmma_f32_16x16x32_f16(false, a, false, b,
                                                   (short)0, acc, false, false);
    }
    // bias + tanh, scatter D tile (M = r + 8*hi, N = n0 + nl) to LDS
    float bias = lin_b[n0 + nl];
    #pragma unroll
    for (int r = 0; r < 8; ++r) {
      int m = r + 8 * hi;
      Otile[m * OPAD + n0 + nl] = (_Float16)tanhf(acc[r] + bias);
    }
  }
  __syncthreads();

  // ---- Phase 3: + type/order/segment embeddings, LayerNorm(eps=1e-12) ----
  {
    const int r = t >> 4, q = t & 15;     // 16 threads per token row
    const int tok = token0 + r;
    const float* Wt = W_type  + (size_t)type_ids[tok]       * Hh;
    const float* Wo = W_order + (size_t)visit_orders[tok]   * Hh;
    const float* Wg = W_seg   + (size_t)visit_segments[tok] * Hh;
    float vv[48];
    float sum = 0.f, sq = 0.f;
    #pragma unroll
    for (int i = 0; i < 48; ++i) {
      int c = q + 16 * i;
      float v = (float)Otile[r * OPAD + c] + Wt[c] + Wo[c] + Wg[c];
      vv[i] = v; sum += v; sq += v * v;
    }
    rsum[t] = sum; rsq[t] = sq;
    __syncthreads();
    float ts = 0.f, tq = 0.f;
    #pragma unroll
    for (int i = 0; i < 16; ++i) { ts += rsum[(r << 4) + i]; tq += rsq[(r << 4) + i]; }
    float mu   = ts * (1.f / (float)Hh);
    float var  = tq * (1.f / (float)Hh) - mu * mu;
    float rstd = rsqrtf(var + 1e-12f);
    float* op = out + (size_t)tok * Hh;
    #pragma unroll
    for (int i = 0; i < 48; ++i) {
      int c = q + 16 * i;
      op[c] = (vv[i] - mu) * rstd * ln_g[c] + ln_beta[c];
    }
  }
}

extern "C" void kernel_launch(void* const* d_in, const int* in_sizes, int n_in,
                              void* d_out, int out_size, void* d_ws, size_t ws_size,
                              hipStream_t stream) {
  const int*   input_ids      = (const int*)  d_in[0];
  const int*   type_ids       = (const int*)  d_in[1];
  const float* time_stamps    = (const float*)d_in[2];
  const float* ages           = (const float*)d_in[3];
  const int*   visit_orders   = (const int*)  d_in[4];
  const int*   visit_segments = (const int*)  d_in[5];
  const float* W_word         = (const float*)d_in[6];
  const float* W_type         = (const float*)d_in[7];
  const float* W_order        = (const float*)d_in[8];
  const float* W_seg          = (const float*)d_in[9];
  const float* time_w         = (const float*)d_in[10];
  const float* time_phi       = (const float*)d_in[11];
  const float* age_w          = (const float*)d_in[12];
  const float* age_phi        = (const float*)d_in[13];
  const float* lin_W          = (const float*)d_in[14];
  const float* lin_b          = (const float*)d_in[15];
  const float* ln_g           = (const float*)d_in[16];
  const float* ln_beta        = (const float*)d_in[17];

  _Float16* wT = (_Float16*)d_ws;         // 768*832*2 = 1.28 MB, L2-resident

  const int wTelems = Hh * KD;
  ehr_prep_wT<<<(wTelems + 255) / 256, 256, 0, stream>>>(lin_W, wT);

  const int n_tokens = in_sizes[0];       // B*S = 65536
  ehr_fused<<<n_tokens / 16, 256, 0, stream>>>(
      input_ids, type_ids, time_stamps, ages, visit_orders, visit_segments,
      W_word, W_type, W_order, W_seg, time_w, time_phi, age_w, age_phi,
      wT, lin_b, ln_g, ln_beta, (float*)d_out);
}
